// SGNet_30932354466385
// MI455X (gfx1250) — matrix operand order repaired and checked by reading, not compiled, not run on hardware
//
#include <hip/hip_runtime.h>
#include <hip/hip_bf16.h>

#define HH   512
#define HQ   128
#define ENCS 8
#define DECS 12
#define NPRED 2
#define BB   512

typedef __attribute__((ext_vector_type(16))) __bf16 v16bf;
typedef __attribute__((ext_vector_type(8)))  float  v8f;

__device__ __forceinline__ unsigned short f2bf(float f) {
    unsigned int u = __float_as_uint(f);
    u += 0x7FFFu + ((u >> 16) & 1u);          // round-to-nearest-even
    return (unsigned short)(u >> 16);
}

// ---------------------------------------------------------------------------
// bf16 WMMA GEMM:  C[MxN] = act( A[MxK](bf16,lda) * W[KxN](bf16) + bias )
// Requirements (all call sites satisfy them): M%32==0, N%64==0, K%64==0.
// Each wave owns a 32x64 C tile: 2 A-frags x 4 B-frags = 8 WMMA per K-step,
// K-loop unrolled x2 -> 8 b128 A loads + 128 u16 B loads then 16 WMMAs per iteration, no
// divergent control flow anywhere near the WMMAs (EXEC stays all-ones).
// A fragment (16-bit A 16x32, ISA 7.12.2): lane m=l&15, half=l>>4,
//   slots 0..7 = K[k0+half*8 .. +7], slots 8..15 = K[k0+16+half*8 .. +7].
// B fragment: lane n=l&15, half=l>>4, same K slots as A, column N=n0+n.
// C layout: vgpr g -> M = m0 + g + 8*half, N = n0 + (l&15).
// ---------------------------------------------------------------------------
template <int RELU, int ACC, int WF32, int WBF16>
__global__ __launch_bounds__(256) void gemm_bf16_wmma(
    const unsigned short* __restrict__ A, int lda,
    const unsigned short* __restrict__ Wm,
    const float* __restrict__ bias,
    float* __restrict__ Cf, unsigned short* __restrict__ Cb,
    int M, int N, int K)
{
    const int wave = (blockIdx.x * 256 + threadIdx.x) >> 5;
    const int lane = threadIdx.x & 31;
    const int tilesN = N >> 6;
    const int tm = wave / tilesN;
    const int tn = wave - tm * tilesN;
    const int m0 = tm << 5;      // 32 rows of C
    const int n0 = tn << 6;      // 64 cols of C
    if (m0 >= M) return;         // wave-uniform grid padding guard

    const int mr   = lane & 15;
    const int half = lane >> 4;

    union U16 { uint4 q[2]; v16bf v; unsigned short s[16]; };

    const unsigned short* a0 = A + (size_t)(m0 + mr) * lda + half * 8;
    const unsigned short* a1 = a0 + (size_t)16 * lda;
    const unsigned short* bp = Wm + (size_t)(half * 8) * N + n0 + mr;

    v8f acc[2][4];
#pragma unroll
    for (int mt = 0; mt < 2; ++mt)
#pragma unroll
        for (int s = 0; s < 4; ++s) acc[mt][s] = v8f{};

    for (int k0 = 0; k0 < K; k0 += 64) {
        U16 fa0, fa1, ga0, ga1, fb[4], gb[4];
        // ---- loads for K sub-step 0 (k0) and 1 (k0+32): one big clause ----
        fa0.q[0] = *(const uint4*)(a0 + k0);
        fa0.q[1] = *(const uint4*)(a0 + k0 + 16);
        fa1.q[0] = *(const uint4*)(a1 + k0);
        fa1.q[1] = *(const uint4*)(a1 + k0 + 16);
        ga0.q[0] = *(const uint4*)(a0 + k0 + 32);
        ga0.q[1] = *(const uint4*)(a0 + k0 + 48);
        ga1.q[0] = *(const uint4*)(a1 + k0 + 32);
        ga1.q[1] = *(const uint4*)(a1 + k0 + 48);
        const unsigned short* b0 = bp + (size_t)k0 * N;
        const unsigned short* b1 = b0 + (size_t)32 * N;
#pragma unroll
        for (int s = 0; s < 4; ++s) {
#pragma unroll
            for (int i = 0; i < 16; ++i) {
                fb[s].s[i] = b0[(size_t)(i + (i & 8)) * N + (s << 4)];
                gb[s].s[i] = b1[(size_t)(i + (i & 8)) * N + (s << 4)];
            }
        }
        // ---- 16 WMMAs, straight-line ----
#pragma unroll
        for (int s = 0; s < 4; ++s) {
            acc[0][s] = __builtin_amdgcn_wmma_f32_16x16x32_bf16(
                false, fa0.v, false, fb[s].v, (short)0, acc[0][s], false, false);
            acc[1][s] = __builtin_amdgcn_wmma_f32_16x16x32_bf16(
                false, fa1.v, false, fb[s].v, (short)0, acc[1][s], false, false);
        }
#pragma unroll
        for (int s = 0; s < 4; ++s) {
            acc[0][s] = __builtin_amdgcn_wmma_f32_16x16x32_bf16(
                false, ga0.v, false, gb[s].v, (short)0, acc[0][s], false, false);
            acc[1][s] = __builtin_amdgcn_wmma_f32_16x16x32_bf16(
                false, ga1.v, false, gb[s].v, (short)0, acc[1][s], false, false);
        }
    }

    // ---- epilogue: bias / accumulate / relu / f32 and bf16 stores ----
#pragma unroll
    for (int s = 0; s < 4; ++s) {
        const int ncol = n0 + (s << 4) + mr;
        const float bv = bias ? bias[ncol] : 0.f;
#pragma unroll
        for (int mt = 0; mt < 2; ++mt) {
#pragma unroll
            for (int g = 0; g < 8; ++g) {
                const int mm = m0 + (mt << 4) + g + (half << 3);
                const size_t off = (size_t)mm * N + ncol;
                float v = acc[mt][s][g] + bv;
                if (ACC)  v += Cf[off];
                if (RELU) v = fmaxf(v, 0.f);
                if (WF32)  Cf[off] = v;
                if (WBF16) Cb[off] = f2bf(v);
            }
        }
    }
}

// ---------------------------------------------------------------------------
__global__ void cvt_f32_bf16(const float* __restrict__ in,
                             unsigned short* __restrict__ out, int n) {
    int i = blockIdx.x * 256 + threadIdx.x;
    if (i < n) out[i] = f2bf(in[i]);
}
__global__ void fill_zero_f32(float* p, int n) {
    int i = blockIdx.x * 256 + threadIdx.x;
    if (i < n) p[i] = 0.f;
}
__global__ void fill_zero_u16(unsigned short* p, int n) {
    int i = blockIdx.x * 256 + threadIdx.x;
    if (i < n) p[i] = 0;
}

// traj_input = relu(inputs @ embed_w + embed_b), K=6 (too small for WMMA)
__global__ void embed_kernel(const float* __restrict__ x,
                             const float* __restrict__ w,
                             const float* __restrict__ b,
                             unsigned short* __restrict__ out, int rows) {
    int i = blockIdx.x * 256 + threadIdx.x;
    if (i >= rows * HH) return;
    int r = i >> 9, c = i & (HH - 1);
    const float* xr = x + r * 6;
    float a = b[c];
#pragma unroll
    for (int k = 0; k < 6; ++k) a += xr[k] * w[k * HH + c];
    out[i] = f2bf(fmaxf(a, 0.f));
}

// h' = (1-z)*n + z*h ; gates split [r|z|n] per reference ordering
__global__ void gru_fuse(const float* __restrict__ gi,
                         const float* __restrict__ gh,
                         float* __restrict__ h,
                         unsigned short* __restrict__ hbf,
                         unsigned short* __restrict__ hbf2,
                         int total, int Hn) {
    int i = blockIdx.x * 256 + threadIdx.x;
    if (i >= total) return;
    int b = i / Hn, j = i - b * Hn;
    const float* gir = gi + (size_t)b * 3 * Hn;
    const float* ghr = gh + (size_t)b * 3 * Hn;
    float r = 1.f / (1.f + expf(-(gir[j]          + ghr[j])));
    float z = 1.f / (1.f + expf(-(gir[Hn + j]     + ghr[Hn + j])));
    float n = tanhf(gir[2 * Hn + j] + r * ghr[2 * Hn + j]);
    float hv = (1.f - z) * n + z * h[i];
    h[i] = hv;
    unsigned short bf = f2bf(hv);
    if (hbf)  hbf[i]  = bf;
    if (hbf2) hbf2[i] = bf;
}

// goal_for_enc: softmax over relu(tanh(gfe)@ea_w+ea_b), weighted sum of gfe
__global__ __launch_bounds__(128) void attn_enc_k(
    const float* __restrict__ gfe_all,   // [DEC,B,HQ] f32
    const float* __restrict__ eaw, const float* __restrict__ eab,
    unsigned short* __restrict__ gfe_out, int Bn) {
    int b = blockIdx.x, j = threadIdx.x;
    __shared__ float red[HQ];
    __shared__ float logits[DECS];
    float w = eaw[j];
    float gv[DECS];
    for (int t = 0; t < DECS; ++t) {
        float g = gfe_all[((size_t)t * Bn + b) * HQ + j];
        gv[t] = g;
        red[j] = tanhf(g) * w;
        __syncthreads();
        for (int off = 64; off > 0; off >>= 1) {
            if (j < off) red[j] += red[j + off];
            __syncthreads();
        }
        if (j == 0) logits[t] = fmaxf(red[0] + eab[0], 0.f);
        __syncthreads();
    }
    float mx = logits[0];
    for (int t = 1; t < DECS; ++t) mx = fmaxf(mx, logits[t]);
    float den = 0.f, at[DECS];
    for (int t = 0; t < DECS; ++t) { at[t] = expf(logits[t] - mx); den += at[t]; }
    float inv = 1.f / den, ctx = 0.f;
    for (int t = 0; t < DECS; ++t) ctx += at[t] * inv * gv[t];
    gfe_out[(size_t)b * HQ + j] = f2bf(ctx);
}

// decoder masked attention: entries t<tcur are zeroed (logit=relu(da_b))
__global__ __launch_bounds__(128) void attn_dec_k(
    const float* __restrict__ gfd,       // [DEC,B,HQ] f32
    const float* __restrict__ daw, const float* __restrict__ dab,
    int tcur, unsigned short* __restrict__ ctx_bf, int Bn) {
    int b = blockIdx.x, j = threadIdx.x;
    __shared__ float red[HQ];
    __shared__ float logits[DECS];
    float w = daw[j];
    float gv[DECS];
    for (int t = 0; t < DECS; ++t) {
        float g = (t >= tcur) ? gfd[((size_t)t * Bn + b) * HQ + j] : 0.f;
        gv[t] = g;
        red[j] = tanhf(g) * w;
        __syncthreads();
        for (int off = 64; off > 0; off >>= 1) {
            if (j < off) red[j] += red[j + off];
            __syncthreads();
        }
        if (j == 0) logits[t] = fmaxf(red[0] + dab[0], 0.f);
        __syncthreads();
    }
    float mx = logits[0];
    for (int t = 1; t < DECS; ++t) mx = fmaxf(mx, logits[t]);
    float den = 0.f, at[DECS];
    for (int t = 0; t < DECS; ++t) { at[t] = expf(logits[t] - mx); den += at[t]; }
    float inv = 1.f / den, ctx = 0.f;
    for (int t = 0; t < DECS; ++t) ctx += at[t] * inv * gv[t];
    ctx_bf[(size_t)b * HQ + j] = f2bf(ctx);
}

// X[m,512] @ reg_w[512,2] + reg_b -> out[((b*ENC+e)*DEC+t)*2 + p]
// mode 0 (goal): m = t*B + b ; mode 1 (dec): b = m, t = tcur
__global__ __launch_bounds__(128) void reg_proj(
    const float* __restrict__ X, const float* __restrict__ w,
    const float* __restrict__ bvec, float* __restrict__ out,
    int mode, int e, int tcur, int Bn) {
    int mrow = blockIdx.x, tid = threadIdx.x;
    const float* x = X + (size_t)mrow * HH;
    float p0 = 0.f, p1 = 0.f;
    for (int k = tid; k < HH; k += 128) {
        float xv = x[k];
        p0 += xv * w[2 * k];
        p1 += xv * w[2 * k + 1];
    }
    __shared__ float s0[128], s1[128];
    s0[tid] = p0; s1[tid] = p1;
    __syncthreads();
    for (int off = 64; off > 0; off >>= 1) {
        if (tid < off) { s0[tid] += s0[tid + off]; s1[tid] += s1[tid + off]; }
        __syncthreads();
    }
    if (tid == 0) {
        int b, t;
        if (mode == 0) { t = mrow / Bn; b = mrow - t * Bn; }
        else           { b = mrow; t = tcur; }
        size_t o = (((size_t)b * ENCS + e) * DECS + t) * NPRED;
        out[o]     = s0[0] + bvec[0];
        out[o + 1] = s1[0] + bvec[1];
    }
}

// ---------------------------------------------------------------------------
enum { GM_F32 = 0, GM_ACC = 1, GM_RELU_F32_BF16 = 2, GM_RELU_F32 = 3, GM_RELU_BF16 = 4 };

static void gemm(hipStream_t st, int mode, const unsigned short* A, int lda,
                 const unsigned short* W, const float* bias,
                 float* Cf, unsigned short* Cb, int M, int N, int K) {
    int tiles = (M >> 5) * (N >> 6);           // 32x64 C tile per wave
    dim3 grid((unsigned)((tiles * 32 + 255) / 256)), blk(256);
    switch (mode) {
    case GM_F32:           hipLaunchKernelGGL((gemm_bf16_wmma<0,0,1,0>), grid, blk, 0, st, A, lda, W, bias, Cf, Cb, M, N, K); break;
    case GM_ACC:           hipLaunchKernelGGL((gemm_bf16_wmma<0,1,1,0>), grid, blk, 0, st, A, lda, W, bias, Cf, Cb, M, N, K); break;
    case GM_RELU_F32_BF16: hipLaunchKernelGGL((gemm_bf16_wmma<1,0,1,1>), grid, blk, 0, st, A, lda, W, bias, Cf, Cb, M, N, K); break;
    case GM_RELU_F32:      hipLaunchKernelGGL((gemm_bf16_wmma<1,0,1,0>), grid, blk, 0, st, A, lda, W, bias, Cf, Cb, M, N, K); break;
    case GM_RELU_BF16:     hipLaunchKernelGGL((gemm_bf16_wmma<1,0,0,1>), grid, blk, 0, st, A, lda, W, bias, Cf, Cb, M, N, K); break;
    }
}

extern "C" void kernel_launch(void* const* d_in, const int* in_sizes, int n_in,
                              void* d_out, int out_size, void* d_ws, size_t ws_size,
                              hipStream_t stream) {
    const float* inputs   = (const float*)d_in[0];
    const float* embed_w  = (const float*)d_in[1];
    const float* embed_b  = (const float*)d_in[2];
    const float* reg_w    = (const float*)d_in[3];
    const float* reg_b    = (const float*)d_in[4];
    const float* ea_w     = (const float*)d_in[5];
    const float* ea_b     = (const float*)d_in[6];
    const float* da_w     = (const float*)d_in[7];
    const float* da_b     = (const float*)d_in[8];
    const float* e2g_w    = (const float*)d_in[9];
    const float* e2g_b    = (const float*)d_in[10];
    const float* e2d_w    = (const float*)d_in[11];
    const float* e2d_b    = (const float*)d_in[12];
    const float* gh2i_w   = (const float*)d_in[13];
    const float* gh2i_b   = (const float*)d_in[14];
    const float* dh2i_w   = (const float*)d_in[15];
    const float* dh2i_b   = (const float*)d_in[16];
    const float* gh2t_w   = (const float*)d_in[17];
    const float* gh2t_b   = (const float*)d_in[18];
    const float* g2e_w    = (const float*)d_in[19];
    const float* g2e_b    = (const float*)d_in[20];
    const float* g2d_w    = (const float*)d_in[21];
    const float* g2d_b    = (const float*)d_in[22];
    const float* enc_wih  = (const float*)d_in[23];
    const float* enc_whh  = (const float*)d_in[24];
    const float* enc_bih  = (const float*)d_in[25];
    const float* enc_bhh  = (const float*)d_in[26];
    const float* goal_wih = (const float*)d_in[27];
    const float* goal_whh = (const float*)d_in[28];
    const float* goal_bih = (const float*)d_in[29];
    const float* goal_bhh = (const float*)d_in[30];
    const float* dec_wih  = (const float*)d_in[31];
    const float* dec_whh  = (const float*)d_in[32];
    const float* dec_bih  = (const float*)d_in[33];
    const float* dec_bhh  = (const float*)d_in[34];

    // ---- workspace bump allocator (256B aligned) ----
    size_t off = 0;
    auto alloc = [&](size_t bytes) -> void* {
        off = (off + 255) & ~(size_t)255;
        void* p = (char*)d_ws + off;
        off += bytes;
        return p;
    };
    auto cvtW = [&](const float* src, int n) -> unsigned short* {
        unsigned short* p = (unsigned short*)alloc((size_t)n * 2);
        hipLaunchKernelGGL(cvt_f32_bf16, dim3((n + 255) / 256), dim3(256), 0, stream, src, p, n);
        return p;
    };

    // bf16 weights (converted every call; cheap vs. 235 GFLOP of GEMM)
    unsigned short* wb_enc_wih  = cvtW(enc_wih,  (HH + HQ) * 3 * HH);
    unsigned short* wb_enc_whh  = cvtW(enc_whh,  HH * 3 * HH);
    unsigned short* wb_dec_wih  = cvtW(dec_wih,  (HH + HQ) * 3 * HH);
    unsigned short* wb_dec_whh  = cvtW(dec_whh,  HH * 3 * HH);
    unsigned short* wb_goal_wih = cvtW(goal_wih, HQ * 3 * HQ);
    unsigned short* wb_goal_whh = cvtW(goal_whh, HQ * 3 * HQ);
    unsigned short* wb_e2g      = cvtW(e2g_w,    HH * HQ);
    unsigned short* wb_e2d      = cvtW(e2d_w,    HH * HH);
    unsigned short* wb_gh2i     = cvtW(gh2i_w,   HQ * HQ);
    unsigned short* wb_dh2i     = cvtW(dh2i_w,   HH * HH);
    unsigned short* wb_gh2t     = cvtW(gh2t_w,   HQ * HH);
    unsigned short* wb_g2e      = cvtW(g2e_w,    HQ * HQ);
    unsigned short* wb_g2d      = cvtW(g2d_w,    HQ * HQ);

    // activations
    unsigned short* traj_bf  = (unsigned short*)alloc((size_t)BB * ENCS * HH * 2);
    float*          teh      = (float*)alloc((size_t)BB * HH * 4);
    unsigned short* teh_bf   = (unsigned short*)alloc((size_t)BB * HH * 2);
    unsigned short* gfe_bf   = (unsigned short*)alloc((size_t)BB * HQ * 2);
    float*          gate_i   = (float*)alloc((size_t)BB * 3 * HH * 4);   // reused enc/goal/dec
    float*          gate_h   = (float*)alloc((size_t)BB * 3 * HH * 4);
    float*          ghid     = (float*)alloc((size_t)BB * HQ * 4);
    unsigned short* ghid_bf  = (unsigned short*)alloc((size_t)BB * HQ * 2);
    float*          dh       = (float*)alloc((size_t)BB * HH * 4);
    unsigned short* dh_bf    = (unsigned short*)alloc((size_t)BB * HH * 2);
    unsigned short* goal_in  = (unsigned short*)alloc((size_t)BB * HQ * 2);
    unsigned short* goals_bf = (unsigned short*)alloc((size_t)DECS * BB * HQ * 2);
    float*          ht       = (float*)alloc((size_t)DECS * BB * HH * 4);
    float*          gfe_all  = (float*)alloc((size_t)DECS * BB * HQ * 4);
    float*          gfd      = (float*)alloc((size_t)DECS * BB * HQ * 4);
    unsigned short* ctx_bf   = (unsigned short*)alloc((size_t)BB * HQ * 2);
    unsigned short* ddi_bf   = (unsigned short*)alloc((size_t)BB * HH * 2);

    float* out_goal = (float*)d_out;
    float* out_dec  = out_goal + (size_t)BB * ENCS * DECS * NPRED;

    // ---- embed + initial state ----
    {
        int n = BB * ENCS * HH;
        hipLaunchKernelGGL(embed_kernel, dim3((n + 255) / 256), dim3(256), 0, stream,
                           inputs, embed_w, embed_b, traj_bf, BB * ENCS);
        hipLaunchKernelGGL(fill_zero_f32, dim3((BB * HH + 255) / 256), dim3(256), 0, stream, teh, BB * HH);
        hipLaunchKernelGGL(fill_zero_u16, dim3((BB * HH + 255) / 256), dim3(256), 0, stream, teh_bf, BB * HH);
        hipLaunchKernelGGL(fill_zero_u16, dim3((BB * HQ + 255) / 256), dim3(256), 0, stream, gfe_bf, BB * HQ);
    }

    for (int e = 0; e < ENCS; ++e) {
        // --- encoder GRU: x = concat(traj[:,e,:], gfe) -> split GEMM ---
        gemm(stream, GM_F32, traj_bf + (size_t)e * HH, ENCS * HH, wb_enc_wih, enc_bih,
             gate_i, nullptr, BB, 3 * HH, HH);
        gemm(stream, GM_ACC, gfe_bf, HQ, wb_enc_wih + (size_t)HH * 3 * HH, nullptr,
             gate_i, nullptr, BB, 3 * HH, HQ);
        gemm(stream, GM_F32, teh_bf, HH, wb_enc_whh, enc_bhh, gate_h, nullptr, BB, 3 * HH, HH);
        hipLaunchKernelGGL(gru_fuse, dim3((BB * HH + 255) / 256), dim3(256), 0, stream,
                           gate_i, gate_h, teh, teh_bf, (unsigned short*)nullptr, BB * HH, HH);

        // goal_hidden / dec_hidden
        gemm(stream, GM_RELU_F32_BF16, teh_bf, HH, wb_e2g, e2g_b, ghid, ghid_bf, BB, HQ, HH);
        gemm(stream, GM_RELU_F32_BF16, teh_bf, HH, wb_e2d, e2d_b, dh, dh_bf, BB, HH, HH);

        // --- SGE: 12-step goal GRU ---
        hipLaunchKernelGGL(fill_zero_u16, dim3((BB * HQ + 255) / 256), dim3(256), 0, stream, goal_in, BB * HQ);
        for (int t = 0; t < DECS; ++t) {
            gemm(stream, GM_F32, goal_in, HQ, wb_goal_wih, goal_bih, gate_i, nullptr, BB, 3 * HQ, HQ);
            gemm(stream, GM_F32, ghid_bf, HQ, wb_goal_whh, goal_bhh, gate_h, nullptr, BB, 3 * HQ, HQ);
            hipLaunchKernelGGL(gru_fuse, dim3((BB * HQ + 255) / 256), dim3(256), 0, stream,
                               gate_i, gate_h, ghid, ghid_bf, goals_bf + (size_t)t * BB * HQ,
                               BB * HQ, HQ);
            gemm(stream, GM_RELU_BF16, ghid_bf, HQ, wb_gh2i, gh2i_b, nullptr, goal_in, BB, HQ, HQ);
        }
        // goal trajectory: relu(goals@gh2t+b) @ reg
        gemm(stream, GM_RELU_F32, goals_bf, HQ, wb_gh2t, gh2t_b, ht, nullptr, DECS * BB, HH, HQ);
        hipLaunchKernelGGL(reg_proj, dim3(DECS * BB), dim3(128), 0, stream,
                           ht, reg_w, reg_b, out_goal, 0, e, 0, BB);
        // goal_for_enc attention (updates gfe for next encoder step)
        gemm(stream, GM_RELU_F32, goals_bf, HQ, wb_g2e, g2e_b, gfe_all, nullptr, DECS * BB, HQ, HQ);
        hipLaunchKernelGGL(attn_enc_k, dim3(BB), dim3(128), 0, stream, gfe_all, ea_w, ea_b, gfe_bf, BB);
        // goal_for_dec
        gemm(stream, GM_RELU_F32, goals_bf, HQ, wb_g2d, g2d_b, gfd, nullptr, DECS * BB, HQ, HQ);

        // --- decoder: 12-step GRU with masked attention ---
        for (int t = 0; t < DECS; ++t) {
            hipLaunchKernelGGL(attn_dec_k, dim3(BB), dim3(128), 0, stream, gfd, da_w, da_b, t, ctx_bf, BB);
            gemm(stream, GM_RELU_BF16, dh_bf, HH, wb_dh2i, dh2i_b, nullptr, ddi_bf, BB, HH, HH);
            // x = concat(ctx, ddi) -> split GEMM over dec_wih rows
            gemm(stream, GM_F32, ctx_bf, HQ, wb_dec_wih, dec_bih, gate_i, nullptr, BB, 3 * HH, HQ);
            gemm(stream, GM_ACC, ddi_bf, HH, wb_dec_wih + (size_t)HQ * 3 * HH, nullptr,
                 gate_i, nullptr, BB, 3 * HH, HH);
            gemm(stream, GM_F32, dh_bf, HH, wb_dec_whh, dec_bhh, gate_h, nullptr, BB, 3 * HH, HH);
            hipLaunchKernelGGL(gru_fuse, dim3((BB * HH + 255) / 256), dim3(256), 0, stream,
                               gate_i, gate_h, dh, dh_bf, (unsigned short*)nullptr, BB * HH, HH);
            hipLaunchKernelGGL(reg_proj, dim3(BB), dim3(128), 0, stream,
                               dh, reg_w, reg_b, out_dec, 1, e, t, BB);
        }
    }
    (void)in_sizes; (void)n_in; (void)out_size; (void)ws_size;
}
